// ActiveConv_71073118814974
// MI455X (gfx1250) — matrix operand order repaired and checked by reading, not compiled, not run on hardware
//
#include <hip/hip_runtime.h>
#include <hip/hip_bf16.h>

// Shapes from the reference
#define Bn    16
#define C_IN  64
#define Hn    64
#define Wn    64
#define OPC   4
#define Cn    256      // C_IN * OPC
#define C_OUT 128
#define HW    (Hn * Wn)   // 4096

typedef __attribute__((ext_vector_type(16))) _Float16 v16h;
typedef __attribute__((ext_vector_type(8)))  _Float16 v8h;
typedef __attribute__((ext_vector_type(8)))  float    v8f;

union HV {
    v16h v16;
    v8h  v8[2];
};

// ---------------------------------------------------------------------------
// Prep 1: conv_w f32 [128][256] -> f16 [128][256] (A-matrix source, L2 resident)
// ---------------------------------------------------------------------------
__global__ void prep_weights(const float* __restrict__ w, _Float16* __restrict__ wh) {
    int i = blockIdx.x * 256 + threadIdx.x;   // grid 128 * 256 = 32768
    wh[i] = (_Float16)w[i];
}

// ---------------------------------------------------------------------------
// Prep 2: per-channel constant displacement -> integer shift + 4 blend weights
// offsets[c] = (dx, dy); floor/frac are constant per channel.
// ---------------------------------------------------------------------------
__global__ void prep_params(const float* __restrict__ offsets,
                            float4* __restrict__ wts, int2* __restrict__ shifts) {
    int c = threadIdx.x;                      // 1 block of 256
    float dx = offsets[2 * c + 0];
    float dy = offsets[2 * c + 1];
    float fx = floorf(dx), fy = floorf(dy);
    float wx = dx - fx,    wy = dy - fy;
    wts[c]    = make_float4((1.f - wy) * (1.f - wx),   // v00
                            (1.f - wy) * wx,           // v01 (x+1)
                            wy * (1.f - wx),           // v10 (y+1)
                            wy * wx);                  // v11 (x+1,y+1)
    shifts[c] = make_int2((int)fx, (int)fy);
}

// ---------------------------------------------------------------------------
// Main: per block = one (image, row) -> 64 pixels.
//   Stage 1: build displaced activation tile sAct[64 px][256 ch] in f16 (LDS).
//   Stage 2: 8 waves, wave wv owns C_OUT rows [16*wv, 16*wv+16):
//            32x v_wmma_f32_16x16x32_f16, f32 accumulate, bias epilogue.
// ---------------------------------------------------------------------------
__global__ __launch_bounds__(256) void active_conv_main(
    const float*    __restrict__ inp,     // [16][64][64][64]
    const _Float16* __restrict__ wh,      // [128][256] f16
    const float*    __restrict__ conv_b,  // [128]
    const float4*   __restrict__ wts,     // [256]
    const int2*     __restrict__ shifts,  // [256]
    float*          __restrict__ out)     // [16][128][64][64]
{
    __shared__ __align__(16) _Float16 sAct[64 * Cn];   // 32 KB: [pixel][channel]

    const int tid = threadIdx.x;
    const int img = blockIdx.x >> 6;       // /64
    const int row = blockIdx.x & 63;       // y

    // ---------------- Stage 1: displaced activation tile ----------------
    {
        const int c   = tid;               // this thread owns channel c for all 64 px
        const int cin = c >> 2;            // repeat(OPC=4)
        const float4 wt = wts[c];
        const int2   sh = shifts[c];
        const float* ip = inp + (((size_t)img * C_IN + cin) << 12);  // *4096
        const int  sy0  = row + sh.y;
        const bool okY0 = (sy0 >= 0)     && (sy0 < Hn);
        const bool okY1 = (sy0 + 1 >= 0) && (sy0 + 1 < Hn);
        const float* r0 = ip + sy0 * Wn;
        const float* r1 = r0 + Wn;
        #pragma unroll 4
        for (int n = 0; n < 64; ++n) {
            int  sx0  = n + sh.x;
            bool okX0 = (sx0 >= 0)     && (sx0 < Wn);
            bool okX1 = (sx0 + 1 >= 0) && (sx0 + 1 < Wn);
            float v00 = (okY0 && okX0) ? r0[sx0]     : 0.f;
            float v01 = (okY0 && okX1) ? r0[sx0 + 1] : 0.f;
            float v10 = (okY1 && okX0) ? r1[sx0]     : 0.f;
            float v11 = (okY1 && okX1) ? r1[sx0 + 1] : 0.f;
            float v = wt.x * v00 + wt.y * v01 + wt.z * v10 + wt.w * v11;
            sAct[n * Cn + c] = (_Float16)v;
        }
    }
    __syncthreads();

    // ---------------- Stage 2: WMMA GEMM  D = W(16x32) x Act(32x16) ----------------
    const int lane = tid & 31;
    const int wv   = tid >> 5;             // 0..7
    const int m0   = wv << 4;              // C_OUT tile base
    const int lrow = lane & 15;
    const int half = lane >> 4;            // 0 or 1

    // A k order chosen here (contiguous, same as B): lanes 0-15 -> K 0..15; lanes 16-31 -> K 16..31
    const int khalfA = half << 4;          // halves
    // B k order chosen here (contiguous, same as A): lanes 0-15 -> K 0..15; lanes 16-31 -> K 16..31
    const int khalfB = half << 4;          // halves

    v8f acc0 = {}, acc1 = {}, acc2 = {}, acc3 = {};

    const _Float16* wrow = wh + (m0 + lrow) * Cn + khalfA;

    #pragma unroll
    for (int k = 0; k < 8; ++k) {          // K = 256 in steps of 32
        HV a;
        const v8h* ap = (const v8h*)(wrow + k * 32);
        a.v8[0] = ap[0];                   // K base .. +7
        a.v8[1] = ap[1];                   // K base+8 .. +15

        const v8h* bp = (const v8h*)(&sAct[lrow * Cn + k * 32 + khalfB]);
        HV b0, b1, b2, b3;                 // pixel n = nt*16 + lrow -> +nt*16*256 halves = +512 v8h
        b0.v8[0] = bp[0];     b0.v8[1] = bp[1];
        b1.v8[0] = bp[512];   b1.v8[1] = bp[513];
        b2.v8[0] = bp[1024];  b2.v8[1] = bp[1025];
        b3.v8[0] = bp[1536];  b3.v8[1] = bp[1537];

        acc0 = __builtin_amdgcn_wmma_f32_16x16x32_f16(false, a.v16, false, b0.v16,
                                                      (short)0, acc0, false, false);
        acc1 = __builtin_amdgcn_wmma_f32_16x16x32_f16(false, a.v16, false, b1.v16,
                                                      (short)0, acc1, false, false);
        acc2 = __builtin_amdgcn_wmma_f32_16x16x32_f16(false, a.v16, false, b2.v16,
                                                      (short)0, acc2, false, false);
        acc3 = __builtin_amdgcn_wmma_f32_16x16x32_f16(false, a.v16, false, b3.v16,
                                                      (short)0, acc3, false, false);
    }

    // ---------------- Epilogue: bias + store ----------------
    // C/D layout: VGPR r, lanes 0-15 -> (M=r, N=lane); lanes 16-31 -> (M=r+8, N=lane-16)
    const int mb = m0 + (half << 3);
    float bias[8];
    #pragma unroll
    for (int r = 0; r < 8; ++r) bias[r] = conv_b[mb + r];

    float* obase = out + (((size_t)img * C_OUT + mb) << 12) + (row << 6) + lrow;
    #pragma unroll
    for (int r = 0; r < 8; ++r) {
        float* op = obase + ((size_t)r << 12);   // next C_OUT row
        op[0]  = acc0[r] + bias[r];
        op[16] = acc1[r] + bias[r];
        op[32] = acc2[r] + bias[r];
        op[48] = acc3[r] + bias[r];
    }
}

// ---------------------------------------------------------------------------
extern "C" void kernel_launch(void* const* d_in, const int* in_sizes, int n_in,
                              void* d_out, int out_size, void* d_ws, size_t ws_size,
                              hipStream_t stream) {
    const float* inp     = (const float*)d_in[0];   // 16*64*64*64
    const float* offsets = (const float*)d_in[1];   // 256*2
    const float* conv_w  = (const float*)d_in[2];   // 128*256
    const float* conv_b  = (const float*)d_in[3];   // 128
    float* out = (float*)d_out;                     // 16*128*64*64

    _Float16* wh   = (_Float16*)d_ws;                         // 64 KB
    float4*   wts  = (float4*)((char*)d_ws + 65536);          // 4 KB
    int2*     shft = (int2*)((char*)d_ws + 65536 + 4096);     // 2 KB

    prep_weights<<<dim3(C_OUT * Cn / 256), dim3(256), 0, stream>>>(conv_w, wh);
    prep_params<<<dim3(1), dim3(256), 0, stream>>>(offsets, wts, shft);
    active_conv_main<<<dim3(Bn * Hn), dim3(256), 0, stream>>>(inp, wh, conv_b, wts, shft, out);
}
